// RTDETRDecoder_74242804679340
// MI455X (gfx1250) — matrix operand order repaired and checked by reading, not compiled, not run on hardware
//
#include <hip/hip_runtime.h>
#include <math.h>

// ---------------- model constants ----------------
#define Dm   256
#define Hh   8
#define DH   32
#define Ll   3
#define Pp   4
#define NCc  80
#define NQq  300
#define NLAY 6
#define FFd  1024
#define Ssz  8400
#define Bsz  16
#define BQ   (Bsz*NQq)          // 4800

// ---------------- WMMA GEMM ----------------
#define BM 128
#define BN 64
#define BK 32
#define LPAD 8                  // pad (halfs) -> 80B rows, conflict-free fragment reads

typedef _Float16 half_t;
typedef _Float16 v4h  __attribute__((ext_vector_type(4)));
typedef _Float16 v8h  __attribute__((ext_vector_type(8)));
typedef _Float16 v16h __attribute__((ext_vector_type(16)));
typedef float    v4f  __attribute__((ext_vector_type(4)));
typedef float    v8f  __attribute__((ext_vector_type(8)));

__device__ __forceinline__ float gelu_f(float x) {
    return 0.5f * x * (1.0f + erff(x * 0.70710678118654752f));
}

// C[m,n] = act( (sum_k A[m,k]*B[k,n]) * scale[n] + bias[n] )
// A[m,k] = A[z*a_bs + m*a_rs + k*a_cs]   (optionally row-masked by rmask[m % rmod])
// B[k,n] = B[k*b_rs + n*b_cs]
// C[m,n] = C[z*c_bs + m*c_rs + n]
// AMODE 0: k-contiguous (a_cs==1). AMODE 1: m-contiguous (a_rs==1).
// BMODE 0: n-contiguous (b_cs==1). BMODE 1: k-contiguous (b_rs==1).
template<int AMODE, int BMODE, bool MASK>
__global__ __launch_bounds__(256) void gemm_f16wmma(
    const float* __restrict__ A, long long a_rs, long long a_cs, long long a_bs,
    const float* __restrict__ Bw, long long b_rs, long long b_cs,
    float* __restrict__ C, long long c_rs, long long c_bs,
    int M, int N, int K,
    const float* __restrict__ bias, const float* __restrict__ scale,
    int act, const float* __restrict__ rmask, int rmod)
{
    __shared__ half_t As [BM][BK + LPAD];   // row-major [m][k]
    __shared__ half_t Bst[BN][BK + LPAD];   // transposed [n][k]

    A += (long long)blockIdx.z * a_bs;
    C += (long long)blockIdx.z * c_bs;
    const int m0 = blockIdx.y * BM;
    const int n0 = blockIdx.x * BN;
    const int tid  = threadIdx.x;
    const int lane = tid & 31, wid = tid >> 5;
    const int wrow = wid >> 1, wcol = wid & 1;     // 4 x 2 wave grid
    const int l16  = lane & 15, hlf = lane >> 4;

    const bool full = (m0 + BM <= M) && (n0 + BN <= N) && ((K & (BK - 1)) == 0);

    v8f acc[2][2];
#pragma unroll
    for (int a = 0; a < 2; ++a)
#pragma unroll
        for (int b = 0; b < 2; ++b)
#pragma unroll
            for (int i = 0; i < 8; ++i) acc[a][b][i] = 0.0f;

    for (int k0 = 0; k0 < K; k0 += BK) {
        if (full) {
            // ---- stage A (128x32): 1024 float4, 4 per thread ----
            if (AMODE == 0) {
#pragma unroll
                for (int j = 0; j < 4; ++j) {
                    int q  = tid + j * 256;
                    int mm = q >> 3, kk = (q & 7) << 2;
                    v4f f = *(const v4f*)(A + (long long)(m0 + mm) * a_rs + (k0 + kk));
                    if (MASK) {
                        float msk = rmask[(m0 + mm) % rmod];
                        f[0] *= msk; f[1] *= msk; f[2] *= msk; f[3] *= msk;
                    }
                    v4h h; h[0] = (half_t)f[0]; h[1] = (half_t)f[1];
                    h[2] = (half_t)f[2]; h[3] = (half_t)f[3];
                    *(v4h*)&As[mm][kk] = h;
                }
                if (k0 + BK < K) {
                    int mm = tid >> 1;
                    __builtin_prefetch(A + (long long)(m0 + mm) * a_rs + (k0 + BK), 0, 3);
                }
            } else {
#pragma unroll
                for (int j = 0; j < 4; ++j) {
                    int q  = tid + j * 256;
                    int kk = q >> 5, mm = (q & 31) << 2;
                    v4f f = *(const v4f*)(A + (long long)(m0 + mm) + (long long)(k0 + kk) * a_cs);
                    if (MASK) {
#pragma unroll
                        for (int i = 0; i < 4; ++i) f[i] *= rmask[(m0 + mm + i) % rmod];
                    }
                    As[mm + 0][kk] = (half_t)f[0];
                    As[mm + 1][kk] = (half_t)f[1];
                    As[mm + 2][kk] = (half_t)f[2];
                    As[mm + 3][kk] = (half_t)f[3];
                }
            }
            // ---- stage B (32x64) transposed: 512 float4, 2 per thread ----
            if (BMODE == 0) {
#pragma unroll
                for (int j = 0; j < 2; ++j) {
                    int q  = tid + j * 256;
                    int kk = q >> 4, nn = (q & 15) << 2;
                    v4f f = *(const v4f*)(Bw + (long long)(k0 + kk) * b_rs + (n0 + nn));
                    Bst[nn + 0][kk] = (half_t)f[0];
                    Bst[nn + 1][kk] = (half_t)f[1];
                    Bst[nn + 2][kk] = (half_t)f[2];
                    Bst[nn + 3][kk] = (half_t)f[3];
                }
            } else {
#pragma unroll
                for (int j = 0; j < 2; ++j) {
                    int q  = tid + j * 256;
                    int nn = q >> 3, kk = (q & 7) << 2;
                    v4f f = *(const v4f*)(Bw + (long long)(k0 + kk) + (long long)(n0 + nn) * b_cs);
                    v4h h; h[0] = (half_t)f[0]; h[1] = (half_t)f[1];
                    h[2] = (half_t)f[2]; h[3] = (half_t)f[3];
                    *(v4h*)&Bst[nn][kk] = h;
                }
            }
        } else {
            // ---- generic guarded edge path ----
#pragma unroll
            for (int i = 0; i < 16; ++i) {
                int idx = tid + i * 256;
                int mm = idx >> 5, kk = idx & 31;
                int gm = m0 + mm, gk = k0 + kk;
                float v = (gm < M && gk < K)
                          ? A[(long long)gm * a_rs + (long long)gk * a_cs] : 0.0f;
                if (MASK && gm < M) v *= rmask[gm % rmod];
                As[mm][kk] = (half_t)v;
            }
#pragma unroll
            for (int i = 0; i < 8; ++i) {
                int idx = tid + i * 256;
                int kk = idx >> 6, nn = idx & 63;
                int gk = k0 + kk, gn = n0 + nn;
                float v = (gk < K && gn < N)
                          ? Bw[(long long)gk * b_rs + (long long)gn * b_cs] : 0.0f;
                Bst[nn][kk] = (half_t)v;
            }
        }
        __syncthreads();

        // ---- fragments: contiguous 2x16B LDS reads per operand ----
        v16h af[2], bf[2];
#pragma unroll
        for (int im = 0; im < 2; ++im) {
            int m = wrow * 32 + im * 16 + l16;
            union { v16h v; v8h h[2]; } u;
            u.h[0] = *(const v8h*)&As[m][hlf * 8];
            u.h[1] = *(const v8h*)&As[m][16 + hlf * 8];
            af[im] = u.v;
        }
#pragma unroll
        for (int in_ = 0; in_ < 2; ++in_) {
            int n = wcol * 32 + in_ * 16 + l16;
            union { v16h v; v8h h[2]; } u;
            u.h[0] = *(const v8h*)&Bst[n][hlf * 8];
            u.h[1] = *(const v8h*)&Bst[n][16 + hlf * 8];
            bf[in_] = u.v;
        }
        // 4 independent WMMAs -> pipeline without hazard NOPs
#pragma unroll
        for (int im = 0; im < 2; ++im)
#pragma unroll
            for (int in_ = 0; in_ < 2; ++in_)
                acc[im][in_] = __builtin_amdgcn_wmma_f32_16x16x32_f16(
                    false, af[im], false, bf[in_], (short)0, acc[im][in_], false, false);
        __syncthreads();
    }

    // ---- epilogue ----
#pragma unroll
    for (int im = 0; im < 2; ++im) {
#pragma unroll
        for (int in_ = 0; in_ < 2; ++in_) {
            int n = n0 + wcol * 32 + in_ * 16 + l16;
#pragma unroll
            for (int r = 0; r < 8; ++r) {
                int m = m0 + wrow * 32 + im * 16 + r + (hlf << 3);
                if (full || (m < M && n < N)) {
                    float v = acc[im][in_][r];
                    if (scale) v *= scale[n];
                    if (bias)  v += bias[n];
                    if (act == 1) v = gelu_f(v);
                    C[(long long)m * c_rs + n] = v;
                }
            }
        }
    }
}

// ---------------- elementwise / misc kernels ----------------
__global__ void k_bnprep(const float* b, const float* g, const float* be,
                         const float* m, const float* v, float* scale, float* shift) {
    int d = blockIdx.x * blockDim.x + threadIdx.x;
    if (d >= Dm) return;
    float sc = g[d] * rsqrtf(v[d] + 1e-5f);
    scale[d] = sc;
    shift[d] = (b[d] - m[d]) * sc + be[d];
}

__global__ void k_anchors(float* anch, float* validf) {
    int s = blockIdx.x * blockDim.x + threadIdx.x;
    if (s >= Ssz) return;
    int hh, ww, rem, lvl;
    if (s < 6400)      { lvl = 0; hh = 80; ww = 80; rem = s; }
    else if (s < 8000) { lvl = 1; hh = 40; ww = 40; rem = s - 6400; }
    else               { lvl = 2; hh = 20; ww = 20; rem = s - 8000; }
    int gy = rem / ww, gx = rem % ww;
    float a[4];
    a[0] = (gx + 0.5f) / (float)ww;
    a[1] = (gy + 0.5f) / (float)hh;
    a[2] = 0.05f * (float)(1 << lvl);
    a[3] = a[2];
    bool valid = true;
    for (int i = 0; i < 4; ++i) valid = valid && (a[i] > 0.01f) && (a[i] < 0.99f);
    for (int i = 0; i < 4; ++i)
        anch[s * 4 + i] = valid ? logf(a[i] / (1.0f - a[i])) : INFINITY;
    validf[s] = valid ? 1.0f : 0.0f;
}

// out = LN(a + b) rowwise (D=256, 256 threads/row); b may be null
__global__ __launch_bounds__(256) void k_add_ln(float* out, const float* a, const float* b,
                                                const float* g, const float* bet) {
    long long row = blockIdx.x;
    int t = threadIdx.x;
    float x = a[row * Dm + t] + (b ? b[row * Dm + t] : 0.0f);
    __shared__ float red[256];
    red[t] = x; __syncthreads();
    for (int s = 128; s > 0; s >>= 1) { if (t < s) red[t] += red[t + s]; __syncthreads(); }
    float mean = red[0] * (1.0f / Dm);
    __syncthreads();
    float d = x - mean;
    red[t] = d * d; __syncthreads();
    for (int s = 128; s > 0; s >>= 1) { if (t < s) red[t] += red[t + s]; __syncthreads(); }
    float var = red[0] * (1.0f / Dm);
    out[row * Dm + t] = d * rsqrtf(var + 1e-5f) * g[t] + bet[t];
}

__global__ void k_add(float* out, const float* a, const float* b, long long n) {
    long long i = (long long)blockIdx.x * blockDim.x + threadIdx.x;
    if (i < n) out[i] = a[i] + b[i];
}

__global__ void k_rowmax(const float* logits, float* scores, long long rows, int cols) {
    long long r = (long long)blockIdx.x * blockDim.x + threadIdx.x;
    if (r >= rows) return;
    const float* p = logits + r * cols;
    float m = p[0];
    for (int c = 1; c < cols; ++c) m = fmaxf(m, p[c]);
    scores[r] = m;
}

__global__ __launch_bounds__(256) void k_topk(float* scores, int* idx) {
    int b = blockIdx.x;
    float* sc = scores + (long long)b * Ssz;
    __shared__ float bv[256];
    __shared__ int   bi[256];
    int t = threadIdx.x;
    for (int j = 0; j < NQq; ++j) {
        float best = -INFINITY; int besti = 0;
        for (int s = t; s < Ssz; s += 256) {
            float v = sc[s];
            if (v > best) { best = v; besti = s; }
        }
        bv[t] = best; bi[t] = besti; __syncthreads();
        for (int st = 128; st > 0; st >>= 1) {
            if (t < st) {
                if (bv[t + st] > bv[t] || (bv[t + st] == bv[t] && bi[t + st] < bi[t])) {
                    bv[t] = bv[t + st]; bi[t] = bi[t + st];
                }
            }
            __syncthreads();
        }
        if (t == 0) { idx[b * NQq + j] = bi[0]; sc[bi[0]] = -INFINITY; }
        __syncthreads();
    }
}

__global__ void k_gather(const float* om, const int* idx, const float* anch,
                         float* target, float* anch_sel) {
    long long i = (long long)blockIdx.x * blockDim.x + threadIdx.x;
    if (i >= (long long)BQ * Dm) return;
    int d = (int)(i & 255);
    long long bq = i >> 8;
    int b = (int)(bq / NQq);
    int q = (int)(bq % NQq);
    int s = idx[b * NQq + q];
    target[i] = om[((long long)b * Ssz + s) * Dm + d];
    if (d < 4) anch_sel[bq * 4 + d] = anch[s * 4 + d];
}

__global__ void k_refinit(const float* bb, const float* anch_sel, float* ref) {
    int i = blockIdx.x * blockDim.x + threadIdx.x;
    if (i >= BQ * 4) return;
    float u = bb[i] + anch_sel[i];
    ref[i] = 1.0f / (1.0f + expf(-u));
}

// MHA: one block per (q, h, b); qh/kh/vh layout (b, n, h*32+d)
__global__ __launch_bounds__(128) void k_attn(const float* qh, const float* kh,
                                              const float* vh, float* o) {
    int q = blockIdx.x, h = blockIdx.y, b = blockIdx.z;
    int t = threadIdx.x;
    __shared__ float sc[NQq];
    __shared__ float red[128];
    const float* qp = qh + ((long long)(b * NQq + q) * Dm) + h * DH;
    for (int k = t; k < NQq; k += 128) {
        const float* kp = kh + ((long long)(b * NQq + k) * Dm) + h * DH;
        float s = 0.0f;
        for (int d = 0; d < DH; ++d) s += qp[d] * kp[d];
        sc[k] = s * 0.176776695296637f;   // 1/sqrt(32)
    }
    __syncthreads();
    float mx = -INFINITY;
    for (int k = t; k < NQq; k += 128) mx = fmaxf(mx, sc[k]);
    red[t] = mx; __syncthreads();
    for (int s = 64; s > 0; s >>= 1) { if (t < s) red[t] = fmaxf(red[t], red[t + s]); __syncthreads(); }
    mx = red[0]; __syncthreads();
    float sm = 0.0f;
    for (int k = t; k < NQq; k += 128) { float e = expf(sc[k] - mx); sc[k] = e; sm += e; }
    red[t] = sm; __syncthreads();
    for (int s = 64; s > 0; s >>= 1) { if (t < s) red[t] += red[t + s]; __syncthreads(); }
    float inv = 1.0f / red[0];
    if (t < DH) {
        float a = 0.0f;
        for (int k = 0; k < NQq; ++k)
            a += sc[k] * vh[((long long)(b * NQq + k) * Dm) + h * DH + t];
        o[((long long)(b * NQq + q) * Dm) + h * DH + t] = a * inv;
    }
}

// softmax over groups of 12 (per head) of the (BQ, 96) attention-logit tensor
__global__ void k_softmax12(float* aw) {
    int i = blockIdx.x * blockDim.x + threadIdx.x;
    if (i >= BQ * Hh) return;
    long long row = i / Hh;
    int h = i % Hh;
    float* p = aw + row * (Hh * Ll * Pp) + h * (Ll * Pp);
    float m = p[0];
    for (int j = 1; j < 12; ++j) m = fmaxf(m, p[j]);
    float s = 0.0f;
    for (int j = 0; j < 12; ++j) { float e = expf(p[j] - m); p[j] = e; s += e; }
    float inv = 1.0f / s;
    for (int j = 0; j < 12; ++j) p[j] *= inv;
}

__device__ __forceinline__ float vsamp(const float* value, long long base,
                                       int x, int y, int W, int H, int ch) {
    if (x < 0 || x >= W || y < 0 || y >= H) return 0.0f;
    return value[(base + (long long)(y * W + x)) * Dm + ch];
}

// MSDA sampling: grid (NQ, B), block 256 = (h=t/32, d=t%32)
__global__ __launch_bounds__(256) void k_msda(const float* value, const float* ref,
                                              const float* off, const float* aw,
                                              float* out) {
    int q = blockIdx.x, b = blockIdx.y;
    int t = threadIdx.x;
    int h = t >> 5;
    long long bq = (long long)b * NQq + q;
    __shared__ float r4[4];
    if (t < 4) r4[t] = ref[bq * 4 + t];
    __syncthreads();
    const float cx = r4[0], cy = r4[1], rw = r4[2], rh = r4[3];
    const int lw[3] = {80, 40, 20};
    const int lo[3] = {0, 6400, 8000};
    float acc = 0.0f;
    for (int l = 0; l < Ll; ++l) {
        int W = lw[l], H = lw[l];
        long long base = (long long)b * Ssz + lo[l];
        for (int p = 0; p < Pp; ++p) {
            int oi = (h * Ll + l) * Pp + p;
            float ox = off[bq * (Hh * Ll * Pp * 2) + oi * 2 + 0];
            float oy = off[bq * (Hh * Ll * Pp * 2) + oi * 2 + 1];
            float a  = aw[bq * (Hh * Ll * Pp) + oi];
            float lx = cx + ox * 0.25f * rw * 0.5f;
            float ly = cy + oy * 0.25f * rh * 0.5f;
            float x = lx * W - 0.5f, y = ly * H - 0.5f;
            float xf = floorf(x), yf = floorf(y);
            int x0 = (int)xf, y0 = (int)yf;
            float fx = x - xf, fy = y - yf;
            float s00 = vsamp(value, base, x0,     y0,     W, H, t);
            float s10 = vsamp(value, base, x0 + 1, y0,     W, H, t);
            float s01 = vsamp(value, base, x0,     y0 + 1, W, H, t);
            float s11 = vsamp(value, base, x0 + 1, y0 + 1, W, H, t);
            acc += a * (s00 * (1 - fx) * (1 - fy) + s10 * fx * (1 - fy)
                      + s01 * (1 - fx) * fy       + s11 * fx * fy);
        }
    }
    out[bq * Dm + t] = acc;
}

__global__ void k_boxupd(const float* bb, float* ref, float* out_bb) {
    int i = blockIdx.x * blockDim.x + threadIdx.x;
    if (i >= BQ * 4) return;
    float prev = fminf(fmaxf(ref[i], 0.0f), 1.0f);
    float iv = logf(fmaxf(prev, 1e-5f)) - logf(fmaxf(1.0f - prev, 1e-5f));
    float v = 1.0f / (1.0f + expf(-(bb[i] + iv)));
    ref[i] = v;
    out_bb[i] = v;
}

// ---------------- host orchestration ----------------
static inline void gemm_any(hipStream_t st, int amode, int bmode, bool mask,
                            const float* A, long long ars, long long acs, long long a_bs,
                            const float* Bw, long long brs, long long bcs,
                            float* C, long long crs, long long cbs,
                            int M, int N, int K, int nb,
                            const float* bias, const float* scale, int act,
                            const float* rmask, int rmod)
{
    dim3 g((N + BN - 1) / BN, (M + BM - 1) / BM, nb);
    if (mask)
        gemm_f16wmma<0, 0, true ><<<g, 256, 0, st>>>(A, ars, acs, a_bs, Bw, brs, bcs, C, crs, cbs, M, N, K, bias, scale, act, rmask, rmod);
    else if (amode == 1 && bmode == 1)
        gemm_f16wmma<1, 1, false><<<g, 256, 0, st>>>(A, ars, acs, a_bs, Bw, brs, bcs, C, crs, cbs, M, N, K, bias, scale, act, rmask, rmod);
    else
        gemm_f16wmma<0, 0, false><<<g, 256, 0, st>>>(A, ars, acs, a_bs, Bw, brs, bcs, C, crs, cbs, M, N, K, bias, scale, act, rmask, rmod);
}

static inline void gemm(hipStream_t st,
                        const float* A, long long ars, long long acs, long long a_bs,
                        const float* Bw, long long brs, long long bcs,
                        float* C, long long crs, long long cbs,
                        int M, int N, int K, int nb,
                        const float* bias, const float* scale, int act,
                        const float* rmask, int rmod)
{
    gemm_any(st, 0, 0, rmask != nullptr, A, ars, acs, a_bs, Bw, brs, bcs,
             C, crs, cbs, M, N, K, nb, bias, scale, act, rmask, rmod);
}

extern "C" void kernel_launch(void* const* d_in, const int* in_sizes, int n_in,
                              void* d_out, int out_size, void* d_ws, size_t ws_size,
                              hipStream_t stream) {
    (void)in_sizes; (void)n_in; (void)out_size; (void)ws_size;
    const float** in = (const float**)d_in;

    // d_in flat map (dict insertion-order flatten of setup_inputs):
    // 0..2 feats; 3..20 proj (3 levels x {w,b,bn_g,bn_b,bn_m,bn_v});
    // 21 enc_w 22 enc_b 23 enc_ln_g 24 enc_ln_b 25 enc_score_w 26 enc_score_b;
    // 27..32 enc_bbox; 33..36 qpos; 37+26*i dec layer i; 193+2*i dec_score; 205+6*i dec_bbox
    const int PROJ = 3, ENC_W = 21, ENC_BBOX = 27, QPOS = 33, DEC0 = 37, DSC0 = 193, DBB0 = 205;

    float* ws = (float*)d_ws;
    size_t cur = 0;
    auto alloc = [&](size_t nfl) { float* p = ws + cur; cur += (nfl + 63) & ~(size_t)63; return p; };

    float* memory  = alloc((size_t)Bsz * Ssz * Dm);   // persistent tokens
    float* om      = alloc((size_t)Bsz * Ssz * Dm);   // enc output; reused as MSDA value
    float* logits  = alloc((size_t)Bsz * Ssz * NCc);
    float* anch    = alloc(Ssz * 4);
    float* validf  = alloc(Ssz);
    float* scores  = alloc((size_t)Bsz * Ssz);
    int*   idx     = (int*)alloc(Bsz * NQq);
    float* anch_sel= alloc(BQ * 4);
    float* outq    = alloc((size_t)BQ * Dm);          // decoder "output"
    float* qpos_   = alloc((size_t)BQ * Dm);
    float* q2      = alloc((size_t)BQ * Dm);
    float* tmp     = alloc((size_t)BQ * Dm);
    float* msda_o  = alloc((size_t)BQ * Dm);
    float* offb    = alloc((size_t)BQ * Hh * Ll * Pp * 2);
    float* awb     = alloc((size_t)BQ * Hh * Ll * Pp);
    float* bb      = alloc(BQ * 4);
    float* refb    = alloc(BQ * 4);
    float* bnscale = alloc(3 * Dm);
    float* bnshift = alloc(3 * Dm);
    float* Sc      = alloc((size_t)BQ * FFd);         // shared scratch (4 x BQ*Dm)
    float* qh = Sc, *kh = Sc + (size_t)BQ * Dm, *vh = Sc + 2ll * BQ * Dm, *attn_o = Sc + 3ll * BQ * Dm;
    float* t1 = Sc, *t2 = Sc + (size_t)BQ * Dm;
    float* ffn_h = Sc, *qpos_h = Sc;

    float* out_bbox   = (float*)d_out;                         // (6,16,300,4)
    float* out_logits = out_bbox + (size_t)NLAY * BQ * 4;      // (6,16,300,80)

    const int lC[3]  = {256, 512, 1024};
    const int lS[3]  = {6400, 1600, 400};
    const int lOf[3] = {0, 6400, 8000};

    // ---- input projection + BN -> memory (b, s, d) ----
    for (int l = 0; l < 3; ++l) {
        const float** pp = in + PROJ + l * 6;
        k_bnprep<<<1, 256, 0, stream>>>(pp[1], pp[2], pp[3], pp[4], pp[5],
                                        bnscale + l * Dm, bnshift + l * Dm);
        // A[m=t,k=c] = feat[b, c, t] (m-contig); B[k=c,n=d] = w[d*C + c] (k-contig)
        gemm_any(stream, 1, 1, false,
                 in[l], 1, lS[l], (long long)lC[l] * lS[l],
                 pp[0], 1, lC[l],
                 memory + (size_t)lOf[l] * Dm, Dm, (long long)Ssz * Dm,
                 lS[l], Dm, lC[l], Bsz, bnshift + l * Dm, bnscale + l * Dm, 0, nullptr, 1);
    }

    k_anchors<<<(Ssz + 255) / 256, 256, 0, stream>>>(anch, validf);

    // ---- encoder: om = LN(mask(memory) @ enc_w + b) ----
    gemm(stream, memory, Dm, 1, 0, in[ENC_W], Dm, 1, om, Dm, 0,
         Bsz * Ssz, Dm, Dm, 1, in[ENC_W + 1], nullptr, 0, validf, Ssz);
    k_add_ln<<<Bsz * Ssz, 256, 0, stream>>>(om, om, nullptr, in[ENC_W + 2], in[ENC_W + 3]);

    // scores = rowmax(om @ enc_score_w + b); top-k; gather
    gemm(stream, om, Dm, 1, 0, in[ENC_W + 4], NCc, 1, logits, NCc, 0,
         Bsz * Ssz, NCc, Dm, 1, in[ENC_W + 5], nullptr, 0, nullptr, 1);
    k_rowmax<<<(Bsz * Ssz + 255) / 256, 256, 0, stream>>>(logits, scores, (long long)Bsz * Ssz, NCc);
    k_topk<<<Bsz, 256, 0, stream>>>(scores, idx);
    k_gather<<<((size_t)BQ * Dm + 255) / 256, 256, 0, stream>>>(om, idx, anch, outq, anch_sel);

    // enc_bbox MLP on the 300 selected rows only
    gemm(stream, outq, Dm, 1, 0, in[ENC_BBOX + 0], Dm, 1, t1, Dm, 0, BQ, Dm, Dm, 1, in[ENC_BBOX + 1], nullptr, 1, nullptr, 1);
    gemm(stream, t1,   Dm, 1, 0, in[ENC_BBOX + 2], Dm, 1, t2, Dm, 0, BQ, Dm, Dm, 1, in[ENC_BBOX + 3], nullptr, 1, nullptr, 1);
    gemm(stream, t2,   Dm, 1, 0, in[ENC_BBOX + 4], 4,  1, bb, 4,  0, BQ, 4,  Dm, 1, in[ENC_BBOX + 5], nullptr, 0, nullptr, 1);
    k_refinit<<<(BQ * 4 + 255) / 256, 256, 0, stream>>>(bb, anch_sel, refb);

    // ---- decoder layers ----
    for (int i = 0; i < NLAY; ++i) {
        const int DB = DEC0 + i * 26;
        // qpos = MLP(ref): (4 -> 512 gelu -> 256)
        gemm(stream, refb,   4,   1, 0, in[QPOS + 0], 2 * Dm, 1, qpos_h, 2 * Dm, 0, BQ, 2 * Dm, 4,      1, in[QPOS + 1], nullptr, 1, nullptr, 1);
        gemm(stream, qpos_h, 2 * Dm, 1, 0, in[QPOS + 2], Dm, 1, qpos_,  Dm,     0, BQ, Dm,     2 * Dm, 1, in[QPOS + 3], nullptr, 0, nullptr, 1);

        // self-attention
        k_add<<<(BQ * Dm + 255) / 256, 256, 0, stream>>>(q2, outq, qpos_, (long long)BQ * Dm);
        gemm(stream, q2,   Dm, 1, 0, in[DB + 0], Dm, 1, qh, Dm, 0, BQ, Dm, Dm, 1, in[DB + 1], nullptr, 0, nullptr, 1);
        gemm(stream, q2,   Dm, 1, 0, in[DB + 2], Dm, 1, kh, Dm, 0, BQ, Dm, Dm, 1, in[DB + 3], nullptr, 0, nullptr, 1);
        gemm(stream, outq, Dm, 1, 0, in[DB + 4], Dm, 1, vh, Dm, 0, BQ, Dm, Dm, 1, in[DB + 5], nullptr, 0, nullptr, 1);
        k_attn<<<dim3(NQq, Hh, Bsz), 128, 0, stream>>>(qh, kh, vh, attn_o);
        gemm(stream, attn_o, Dm, 1, 0, in[DB + 6], Dm, 1, tmp, Dm, 0, BQ, Dm, Dm, 1, in[DB + 7], nullptr, 0, nullptr, 1);
        k_add_ln<<<BQ, 256, 0, stream>>>(outq, outq, tmp, in[DB + 8], in[DB + 9]);

        // MSDA cross-attention (value buffer reuses om)
        float* value = om;
        gemm(stream, memory, Dm, 1, 0, in[DB + 10], Dm, 1, value, Dm, 0,
             Bsz * Ssz, Dm, Dm, 1, in[DB + 11], nullptr, 0, nullptr, 1);
        k_add<<<(BQ * Dm + 255) / 256, 256, 0, stream>>>(q2, outq, qpos_, (long long)BQ * Dm);
        gemm(stream, q2, Dm, 1, 0, in[DB + 12], Hh * Ll * Pp * 2, 1, offb, Hh * Ll * Pp * 2, 0,
             BQ, Hh * Ll * Pp * 2, Dm, 1, in[DB + 13], nullptr, 0, nullptr, 1);
        gemm(stream, q2, Dm, 1, 0, in[DB + 14], Hh * Ll * Pp, 1, awb, Hh * Ll * Pp, 0,
             BQ, Hh * Ll * Pp, Dm, 1, in[DB + 15], nullptr, 0, nullptr, 1);
        k_softmax12<<<(BQ * Hh + 255) / 256, 256, 0, stream>>>(awb);
        k_msda<<<dim3(NQq, Bsz), 256, 0, stream>>>(value, refb, offb, awb, msda_o);
        gemm(stream, msda_o, Dm, 1, 0, in[DB + 16], Dm, 1, tmp, Dm, 0, BQ, Dm, Dm, 1, in[DB + 17], nullptr, 0, nullptr, 1);
        k_add_ln<<<BQ, 256, 0, stream>>>(outq, outq, tmp, in[DB + 18], in[DB + 19]);

        // FFN
        gemm(stream, outq,  Dm,  1, 0, in[DB + 20], FFd, 1, ffn_h, FFd, 0, BQ, FFd, Dm,  1, in[DB + 21], nullptr, 1, nullptr, 1);
        gemm(stream, ffn_h, FFd, 1, 0, in[DB + 22], Dm,  1, tmp,   Dm,  0, BQ, Dm,  FFd, 1, in[DB + 23], nullptr, 0, nullptr, 1);
        k_add_ln<<<BQ, 256, 0, stream>>>(outq, outq, tmp, in[DB + 24], in[DB + 25]);

        // heads
        const int BB = DBB0 + i * 6, SB = DSC0 + i * 2;
        gemm(stream, outq, Dm, 1, 0, in[BB + 0], Dm, 1, t1, Dm, 0, BQ, Dm, Dm, 1, in[BB + 1], nullptr, 1, nullptr, 1);
        gemm(stream, t1,   Dm, 1, 0, in[BB + 2], Dm, 1, t2, Dm, 0, BQ, Dm, Dm, 1, in[BB + 3], nullptr, 1, nullptr, 1);
        gemm(stream, t2,   Dm, 1, 0, in[BB + 4], 4,  1, bb, 4,  0, BQ, 4,  Dm, 1, in[BB + 5], nullptr, 0, nullptr, 1);
        gemm(stream, outq, Dm, 1, 0, in[SB + 0], NCc, 1, out_logits + (size_t)i * BQ * NCc, NCc, 0,
             BQ, NCc, Dm, 1, in[SB + 1], nullptr, 0, nullptr, 1);
        k_boxupd<<<(BQ * 4 + 255) / 256, 256, 0, stream>>>(bb, refb, out_bbox + (size_t)i * BQ * 4);
    }
}
